// TriangleMultiplication_75900662055647
// MI455X (gfx1250) — matrix operand order repaired and checked by reading, not compiled, not run on hardware
//
#include <hip/hip_runtime.h>
#include <hip/hip_fp16.h>

typedef __attribute__((ext_vector_type(16))) _Float16 v16h;
typedef __attribute__((ext_vector_type(8)))  _Float16 v8h;
typedef __attribute__((ext_vector_type(8)))  float    v8f;
typedef _Float16 half_t;

#define NDIM 512
#define CDIM 128
#define HDIM 128
#define NN   (NDIM * NDIM)   // 262144 positions
#define WROWS 768            // p_in(256) | g_in(256) | g_out(128) | p_out(128)

// fast sigmoid: v_exp_f32 + v_rcp_f32 (no IEEE div chain)
__device__ __forceinline__ float sigm(float x) {
  return __builtin_amdgcn_rcpf(1.0f + __expf(-x));
}

// ---- WMMA fragment loaders (contiguous k order, the same for A and B) ----
// A (16x32 f16): lanes 0-15 hold M=lane, K = 0..15; lanes 16-31 hold
// M=lane-16, K = 16..31. Source is row-major [M][K], leading dim ld.
__device__ __forceinline__ v16h load_a_frag(const half_t* base, int ld) {
  int lane = threadIdx.x & 31;
  int m  = lane & 15;
  int hs = (lane >> 4) * 16;
  const half_t* r = base + m * ld;
  v16h a;
#pragma unroll
  for (int e = 0; e < 8; ++e) a[e] = r[hs + e];
#pragma unroll
  for (int e = 0; e < 8; ++e) a[8 + e] = r[hs + 8 + e];
  return a;
}

// B (32x16 f16, computing A.B^T): lane holds column n = lane&15; lanes 0-15
// carry K=0..15, lanes 16-31 carry K=16..31. Source is row-major [N][K].
__device__ __forceinline__ v16h load_b_frag(const half_t* base, int ld) {
  int lane = threadIdx.x & 31;
  int n  = lane & 15;
  int ks = (lane >> 4) * 16;
  const half_t* r = base + n * ld + ks;
  v16h b;
#pragma unroll
  for (int e = 0; e < 16; ++e) b[e] = r[e];
  return b;
}

__device__ __forceinline__ v8f wmma_f16(v16h a, v16h b, v8f c) {
  return __builtin_amdgcn_wmma_f32_16x16x32_f16(false, a, false, b, (short)0, c,
                                                false, false);
}

// ============================================================= kernel 0
// Pack p_in_w(256x128), g_in_w(256x128), g_out_w(128x128), p_out_w(128x128)
// into one half table Wh[768][128], K-contiguous rows.
__global__ __launch_bounds__(256) void k0_cvt(const float* __restrict__ p_in,
                                              const float* __restrict__ g_in,
                                              const float* __restrict__ g_out,
                                              const float* __restrict__ p_out,
                                              half_t* __restrict__ Wh) {
  int idx = blockIdx.x * 256 + threadIdx.x;
  if (idx >= WROWS * CDIM) return;
  int row = idx >> 7;
  int col = idx & 127;
  float v;
  if (row < 256)      v = p_in[idx];
  else if (row < 512) v = g_in[(row - 256) * CDIM + col];
  else if (row < 640) v = g_out[(row - 512) * CDIM + col];
  else                v = p_out[(row - 640) * CDIM + col];
  Wh[idx] = (half_t)v;
}

// ============================================================= kernel 1
// 64 positions per block. LayerNorm(C) in-wave, then WMMA GEMM against Wh
// with weight (B) fragments hoisted across the 4 row-tiles:
//   aT[d][p], bT[d][p]  = h = (xn@p_in^T) * sigmoid(xn@g_in^T)   (channel-major, half)
//   gate[c][p]          = sigmoid(xn@g_out^T)                    (channel-major, half)
#define XN_LD 136
__global__ __launch_bounds__(256) void k1_ln_proj(
    const float* __restrict__ x, const float* __restrict__ nin_w,
    const float* __restrict__ nin_b, const half_t* __restrict__ Wh,
    half_t* __restrict__ aTh, half_t* __restrict__ bTh,
    half_t* __restrict__ gateh) {
  __shared__ half_t xnh[64 * XN_LD];
  const int rowBase0 = blockIdx.x * 64;
  const int tid = threadIdx.x;

  // ---- LayerNorm over C=128: 4 lanes per row, 32 elems per lane ----
  {
    int r  = tid >> 2;          // 0..63
    int c0 = (tid & 3) * 32;
    const float* xr = x + (size_t)(rowBase0 + r) * CDIM + c0;
    float v[32], s = 0.f, sq = 0.f;
#pragma unroll
    for (int e = 0; e < 32; ++e) { v[e] = xr[e]; s += v[e]; sq += v[e] * v[e]; }
    s  += __shfl_xor(s, 1, 32);  sq += __shfl_xor(sq, 1, 32);
    s  += __shfl_xor(s, 2, 32);  sq += __shfl_xor(sq, 2, 32);
    float mean = s * (1.f / CDIM);
    float var  = sq * (1.f / CDIM) - mean * mean;
    float rstd = rsqrtf(var + 1e-5f);
#pragma unroll
    for (int e = 0; e < 32; ++e) {
      float xnv = (v[e] - mean) * rstd * nin_w[c0 + e] + nin_b[c0 + e];
      xnh[r * XN_LD + c0 + e] = (half_t)xnv;
    }
  }
  __syncthreads();

  // ---- projections: 24 column-units x 4 row-tiles ----
  const int wave = tid >> 5;
  const int lane = tid & 31;
  const int n    = lane & 15;
  const int mOff = (lane >> 4) * 8;

  for (int uu = 0; uu < 3; ++uu) {
    int u = wave * 3 + uu;
    if (u < 16) {
      v16h bP[4], bG[4];
#pragma unroll
      for (int kc = 0; kc < 4; ++kc) {
        bP[kc] = load_b_frag(Wh + (size_t)(u * 16) * CDIM + kc * 32, CDIM);
        bG[kc] = load_b_frag(Wh + (size_t)(256 + u * 16) * CDIM + kc * 32, CDIM);
      }
      int d = u * 16 + n;
      half_t* dcol = (d < HDIM) ? (aTh + (size_t)d * NN)
                                : (bTh + (size_t)(d - HDIM) * NN);
#pragma unroll
      for (int rt = 0; rt < 4; ++rt) {
        v8f accP = {}, accG = {};
#pragma unroll
        for (int kc = 0; kc < 4; ++kc) {
          v16h af = load_a_frag(xnh + (rt * 16) * XN_LD + kc * 32, XN_LD);
          accP = wmma_f16(af, bP[kc], accP);
          accG = wmma_f16(af, bG[kc], accG);
        }
        v8h hv;
#pragma unroll
        for (int e = 0; e < 8; ++e) hv[e] = (half_t)(accP[e] * sigm(accG[e]));
        *(v8h*)(dcol + rowBase0 + rt * 16 + mOff) = hv;
      }
    } else {
      int gt = u - 16;
      v16h bF[4];
#pragma unroll
      for (int kc = 0; kc < 4; ++kc)
        bF[kc] = load_b_frag(Wh + (size_t)(512 + gt * 16) * CDIM + kc * 32, CDIM);
      int c = gt * 16 + n;
      half_t* gcol = gateh + (size_t)c * NN;
#pragma unroll
      for (int rt = 0; rt < 4; ++rt) {
        v8f acc = {};
#pragma unroll
        for (int kc = 0; kc < 4; ++kc) {
          v16h af = load_a_frag(xnh + (rt * 16) * XN_LD + kc * 32, XN_LD);
          acc = wmma_f16(af, bF[kc], acc);
        }
        v8h gv;
#pragma unroll
        for (int e = 0; e < 8; ++e) gv[e] = (half_t)sigm(acc[e]);
        *(v8h*)(gcol + rowBase0 + rt * 16 + mOff) = gv;
      }
    }
  }
}

// ============================================================= kernel 2
// Triangle einsum: per channel d, T_d = A_d (512x512) @ B_d^T.
// 128x128 output tile per block; 8 waves x (2m x 4n) WMMA tiles.
// Double-buffered LDS: global loads for chunk k+1 issue before the 8 WMMAs of
// chunk k (latency hidden behind matrix math); one barrier per K-chunk.
#define K2_LD 40
__global__ __launch_bounds__(256) void k2_tri(const half_t* __restrict__ aTh,
                                              const half_t* __restrict__ bTh,
                                              half_t* __restrict__ tH) {
  __shared__ half_t As[2][128 * K2_LD];
  __shared__ half_t Bs[2][128 * K2_LD];
  const int d     = blockIdx.z;
  const int iBase = blockIdx.x * 128;
  const int jBase = blockIdx.y * 128;
  const half_t* A = aTh + (size_t)d * NN;
  const half_t* B = bTh + (size_t)d * NN;
  const int tid  = threadIdx.x;
  const int wave = tid >> 5;
  const int mb   = (wave & 3) * 2;   // m-tiles {mb, mb+1}
  const int nb   = (wave >> 2) * 4;  // n-tiles {nb..nb+3}
  const int lr   = tid >> 1;         // 0..127
  const int lc   = (tid & 1) * 16;   // halves

  const half_t* ga = A + (size_t)(iBase + lr) * NDIM + lc;
  const half_t* gb = B + (size_t)(jBase + lr) * NDIM + lc;
  const int lds_off = lr * K2_LD + lc;

  // prologue: stage chunk 0 into buffer 0
  *(v8h*)(As[0] + lds_off)     = *(const v8h*)(ga);
  *(v8h*)(As[0] + lds_off + 8) = *(const v8h*)(ga + 8);
  *(v8h*)(Bs[0] + lds_off)     = *(const v8h*)(gb);
  *(v8h*)(Bs[0] + lds_off + 8) = *(const v8h*)(gb + 8);
  __syncthreads();

  v8f acc[2][4] = {};
  for (int kc = 0; kc < 16; ++kc) {
    const int cur = kc & 1;
    const int nxt = cur ^ 1;

    // issue global loads for chunk kc+1 (no wait yet)
    v8h ra0, ra1, rb0, rb1;
    if (kc < 15) {
      int k1 = (kc + 1) * 32;
      ra0 = *(const v8h*)(ga + k1);
      ra1 = *(const v8h*)(ga + k1 + 8);
      rb0 = *(const v8h*)(gb + k1);
      rb1 = *(const v8h*)(gb + k1 + 8);
      if (kc < 14) {  // global_prefetch_b8 for chunk kc+2
        __builtin_prefetch(ga + k1 + 32, 0, 0);
        __builtin_prefetch(gb + k1 + 32, 0, 0);
      }
    }

    // compute chunk kc from LDS[cur] while those loads are in flight
    v16h a0 = load_a_frag(As[cur] + (mb * 16) * K2_LD, K2_LD);
    v16h a1 = load_a_frag(As[cur] + (mb * 16 + 16) * K2_LD, K2_LD);
    v16h b0 = load_b_frag(Bs[cur] + (nb + 0) * 16 * K2_LD, K2_LD);
    v16h b1 = load_b_frag(Bs[cur] + (nb + 1) * 16 * K2_LD, K2_LD);
    v16h b2 = load_b_frag(Bs[cur] + (nb + 2) * 16 * K2_LD, K2_LD);
    v16h b3 = load_b_frag(Bs[cur] + (nb + 3) * 16 * K2_LD, K2_LD);
    acc[0][0] = wmma_f16(a0, b0, acc[0][0]);
    acc[0][1] = wmma_f16(a0, b1, acc[0][1]);
    acc[0][2] = wmma_f16(a0, b2, acc[0][2]);
    acc[0][3] = wmma_f16(a0, b3, acc[0][3]);
    acc[1][0] = wmma_f16(a1, b0, acc[1][0]);
    acc[1][1] = wmma_f16(a1, b1, acc[1][1]);
    acc[1][2] = wmma_f16(a1, b2, acc[1][2]);
    acc[1][3] = wmma_f16(a1, b3, acc[1][3]);

    // drain chunk kc+1 into the alternate buffer
    if (kc < 15) {
      *(v8h*)(As[nxt] + lds_off)     = ra0;
      *(v8h*)(As[nxt] + lds_off + 8) = ra1;
      *(v8h*)(Bs[nxt] + lds_off)     = rb0;
      *(v8h*)(Bs[nxt] + lds_off + 8) = rb1;
      __syncthreads();
    }
  }

  const int lane = tid & 31;
  const int n    = lane & 15;
  const int mOff = (lane >> 4) * 8;
  const size_t base = (size_t)d * NN;
#pragma unroll
  for (int mi = 0; mi < 2; ++mi) {
#pragma unroll
    for (int q = 0; q < 4; ++q) {
      int jj = jBase + (nb + q) * 16 + n;
      int i0 = iBase + (mb + mi) * 16 + mOff;
#pragma unroll
      for (int e = 0; e < 8; ++e)
        tH[base + (size_t)(i0 + e) * NDIM + jj] = (half_t)acc[mi][q][e];
    }
  }
}

// ============================================================= kernel 3
// LayerNorm over d of t, output projection (WMMA vs p_out rows of Wh), gate.
#define TF_LD 129
#define TH_LD 136
__global__ __launch_bounds__(256) void k3_out(
    const half_t* __restrict__ tH, const float* __restrict__ nout_w,
    const float* __restrict__ nout_b, const half_t* __restrict__ Wh,
    const half_t* __restrict__ gateh, float* __restrict__ out) {
  __shared__ float  tf[64 * TF_LD];
  __shared__ half_t th[64 * TH_LD];
  const int i     = blockIdx.y;
  const int jBase = blockIdx.x * 64;
  const int tid   = threadIdx.x;

  // load t[d][i][jBase+j] (coalesced over j) into LDS transposed tile tf[j][d]
  {
    int j  = tid & 63;
    int dq = tid >> 6;  // 0..3
#pragma unroll 4
    for (int it = 0; it < 32; ++it) {
      int d = dq * 32 + it;
      tf[j * TF_LD + d] =
          (float)tH[(size_t)d * NN + (size_t)i * NDIM + jBase + j];
    }
  }
  __syncthreads();

  // per-j LayerNorm over d=128: 4 lanes per j, 32 d each
  {
    int j = tid >> 2;
    int q = tid & 3;
    float s = 0.f, sq = 0.f;
#pragma unroll 8
    for (int dd = 0; dd < 32; ++dd) {
      float v = tf[j * TF_LD + q * 32 + dd];
      s += v; sq += v * v;
    }
    s  += __shfl_xor(s, 1, 32);  sq += __shfl_xor(sq, 1, 32);
    s  += __shfl_xor(s, 2, 32);  sq += __shfl_xor(sq, 2, 32);
    float mean = s * (1.f / HDIM);
    float var  = sq * (1.f / HDIM) - mean * mean;
    float rstd = rsqrtf(var + 1e-5f);
#pragma unroll 8
    for (int dd = 0; dd < 32; ++dd) {
      int d = q * 32 + dd;
      float v = tf[j * TF_LD + d];
      th[j * TH_LD + d] = (half_t)((v - mean) * rstd * nout_w[d] + nout_b[d]);
    }
  }
  __syncthreads();

  // GEMM: on (64xj,128d) @ p_out^T (rows 640..767 of Wh), gate, store f32
  const int wave = tid >> 5;
  const int lane = tid & 31;
  const int n    = lane & 15;
  const int mOff = (lane >> 4) * 8;
  const int mt   = wave >> 1;
  const int ntq  = (wave & 1) * 4;

  v16h afr[4];
#pragma unroll
  for (int kc = 0; kc < 4; ++kc)
    afr[kc] = load_a_frag(th + mt * 16 * TH_LD + kc * 32, TH_LD);

  const int jloc = mt * 16 + mOff;
#pragma unroll
  for (int q = 0; q < 4; ++q) {
    int c = (ntq + q) * 16 + n;
    v8f acc = {};
#pragma unroll
    for (int kc = 0; kc < 4; ++kc) {
      v16h bf = load_b_frag(Wh + (size_t)(640 + (ntq + q) * 16) * CDIM + kc * 32,
                            CDIM);
      acc = wmma_f16(afr[kc], bf, acc);
    }
    v8h g8 = *(const v8h*)(gateh + (size_t)c * NN + (size_t)i * NDIM + jBase + jloc);
#pragma unroll
    for (int e = 0; e < 8; ++e) {
      int jj = jBase + jloc + e;
      out[((size_t)i * NDIM + jj) * CDIM + c] = acc[e] * (float)g8[e];
    }
  }
}

// ============================================================= launcher
extern "C" void kernel_launch(void* const* d_in, const int* in_sizes, int n_in,
                              void* d_out, int out_size, void* d_ws,
                              size_t ws_size, hipStream_t stream) {
  const float* x       = (const float*)d_in[0];
  const float* nin_w   = (const float*)d_in[1];
  const float* nin_b   = (const float*)d_in[2];
  const float* p_in_w  = (const float*)d_in[3];
  const float* g_in_w  = (const float*)d_in[4];
  const float* nout_w  = (const float*)d_in[5];
  const float* nout_b  = (const float*)d_in[6];
  const float* p_out_w = (const float*)d_in[7];
  const float* g_out_w = (const float*)d_in[8];
  float* out = (float*)d_out;

  char* ws = (char*)d_ws;
  auto alignup = [](size_t v) { return (v + 255) & ~(size_t)255; };
  half_t* Wh    = (half_t*)ws; ws += alignup((size_t)WROWS * CDIM * sizeof(half_t));
  half_t* aTh   = (half_t*)ws; ws += alignup((size_t)HDIM * NN * sizeof(half_t));
  half_t* bTh   = (half_t*)ws; ws += alignup((size_t)HDIM * NN * sizeof(half_t));
  half_t* gateh = (half_t*)ws; ws += alignup((size_t)CDIM * NN * sizeof(half_t));
  half_t* tH    = (half_t*)ws; ws += alignup((size_t)HDIM * NN * sizeof(half_t));

  k0_cvt<<<(WROWS * CDIM + 255) / 256, 256, 0, stream>>>(p_in_w, g_in_w,
                                                         g_out_w, p_out_w, Wh);
  k1_ln_proj<<<NN / 64, 256, 0, stream>>>(x, nin_w, nin_b, Wh, aTh, bTh, gateh);
  k2_tri<<<dim3(NDIM / 128, NDIM / 128, HDIM), 256, 0, stream>>>(aTh, bTh, tH);
  k3_out<<<dim3(NDIM / 64, NDIM), 256, 0, stream>>>(tH, nout_w, nout_b, Wh,
                                                    gateh, out);
}
